// DAGERC_multimodal_82308753260922
// MI455X (gfx1250) — compile-verified, matrix operand order checked by reading, not run on hardware
//
#include <hip/hip_runtime.h>
#include <hip/hip_bf16.h>
#include <math.h>

// Problem constants (match reference)
#define BB   128
#define NU   100
#define TD   768
#define FOD  512
#define HID  300
#define NCLS 7
#define LNUM 2

typedef __attribute__((ext_vector_type(16))) _Float16 v16h;
typedef __attribute__((ext_vector_type(8)))  float    v8f;

__device__ __forceinline__ float sigm(float x) { return 1.0f / (1.0f + __expf(-x)); }

// Pack two contiguous 8-float runs (K = base..base+7 and base+16..base+23)
// into the 16x16x32 f16 A/B per-lane fragment layout.
__device__ __forceinline__ v16h pack16(float4 a0, float4 a1, float4 a2, float4 a3)
{
    v16h r;
    r[0]  = (_Float16)a0.x; r[1]  = (_Float16)a0.y; r[2]  = (_Float16)a0.z; r[3]  = (_Float16)a0.w;
    r[4]  = (_Float16)a1.x; r[5]  = (_Float16)a1.y; r[6]  = (_Float16)a1.z; r[7]  = (_Float16)a1.w;
    r[8]  = (_Float16)a2.x; r[9]  = (_Float16)a2.y; r[10] = (_Float16)a2.z; r[11] = (_Float16)a2.w;
    r[12] = (_Float16)a3.x; r[13] = (_Float16)a3.y; r[14] = (_Float16)a3.z; r[15] = (_Float16)a3.w;
    return r;
}

__device__ __forceinline__ v16h pack16_ptr(const float* __restrict__ p)
{
    const float4* q = (const float4*)p;
    return pack16(q[0], q[1], q[4], q[5]);
}

// Branch-free bounded fragment load for the K remainder: clamp index, select 0.
__device__ __forceinline__ v16h pack16_bounded(const float* __restrict__ row,
                                               int k0, int koff, int Kdim)
{
    v16h r;
#pragma unroll
    for (int e = 0; e < 16; ++e) {
        const int k  = k0 + koff + (e & 7) + ((e >> 3) << 4);
        const int kc = (k < Kdim) ? k : 0;        // unconditional load from valid addr
        const float v = row[kc];
        r[e] = (_Float16)((k < Kdim) ? v : 0.0f); // v_cndmask, no exec masking
    }
    return r;
}

// ---------------------------------------------------------------------------
// Generic WMMA GEMM:  Y[M,N] = act( X[M,K] @ W[N,K]^T + bias + addin ) * mul
//   X row-major stride ldx; W row-major (N rows of K) stride ldw;
//   addin/mul/Y share layout stride ldy. act: 0=none 1=sigmoid 2=relu.
//   M multiple of 32 (true for all call sites). One wave computes a 32x64
//   strip (2 M-tiles x 4 N-tiles); A and B fragments are each reused, the
//   K-loop is fully branch-free (out-of-range N-tiles compute on a clamped
//   valid W row and are dropped by the lane store guard).
// ---------------------------------------------------------------------------
__global__ void wmma_gemm_kernel(const float* __restrict__ X, int ldx,
                                 const float* __restrict__ W, int ldw,
                                 const float* __restrict__ bias,
                                 const float* __restrict__ addin,
                                 const float* __restrict__ mul,
                                 float* __restrict__ Y, int ldy,
                                 int Mrows, int Kdim, int Ncols, int act)
{
    const int tilesN  = (Ncols + 15) >> 4;
    const int groupsN = (tilesN + 3) >> 2;        // 4 N-tiles per wave
    const int groupsM = Mrows >> 5;               // 2 M-tiles per wave
    // Force wave-uniform scalar: tm/tg comparisons become s_cbranch, not exec.
    const int wave = __builtin_amdgcn_readfirstlane(
        (int)((blockIdx.x * blockDim.x + threadIdx.x) >> 5));
    if (wave >= groupsM * groupsN) return;        // scalar-uniform exit
    const int tmg  = wave / groupsN;
    const int tg   = wave - tmg * groupsN;
    const int lane = threadIdx.x & 31;
    const int l15  = lane & 15;
    const int koff = (lane >> 4) << 3;            // lanes 16-31 hold the +8 K-halves

    const int mrow0 = (tmg << 5) + l15;           // A tile 0: lane -> row M
    const float* __restrict__ xrow0 = X + (size_t)mrow0 * (size_t)ldx;
    const float* __restrict__ xrow1 = xrow0 + (size_t)16 * (size_t)ldx;

    const float* wrow[4];
    int ncolt[4];
#pragma unroll
    for (int t = 0; t < 4; ++t) {
        int nc = (((tg << 2) + t) << 4) + l15;
        ncolt[t] = nc;
        if (nc > Ncols - 1) nc = Ncols - 1;       // clamp: valid row, garbage cols
        wrow[t] = W + (size_t)nc * (size_t)ldw;   //   never stored (guard below)
    }

    v8f acc[2][4];
    {
        v8f z = {};
#pragma unroll
        for (int mi = 0; mi < 2; ++mi)
#pragma unroll
            for (int t = 0; t < 4; ++t) acc[mi][t] = z;
    }

    int k0 = 0;
    for (; k0 + 32 <= Kdim; k0 += 32) {           // full K chunks: b128 loads only
        const v16h a0 = pack16_ptr(xrow0 + k0 + koff);
        const v16h a1 = pack16_ptr(xrow1 + k0 + koff);
#pragma unroll
        for (int t = 0; t < 4; ++t) {
            const v16h b = pack16_ptr(wrow[t] + k0 + koff);
            acc[0][t] = __builtin_amdgcn_wmma_f32_16x16x32_f16(
                            false, a0, false, b, (short)0, acc[0][t], false, false);
            acc[1][t] = __builtin_amdgcn_wmma_f32_16x16x32_f16(
                            false, a1, false, b, (short)0, acc[1][t], false, false);
        }
    }
    if (k0 < Kdim) {                              // K remainder (K=300, 900, ...)
        const v16h a0 = pack16_bounded(xrow0, k0, koff, Kdim);
        const v16h a1 = pack16_bounded(xrow1, k0, koff, Kdim);
#pragma unroll
        for (int t = 0; t < 4; ++t) {
            const v16h b = pack16_bounded(wrow[t], k0, koff, Kdim);
            acc[0][t] = __builtin_amdgcn_wmma_f32_16x16x32_f16(
                            false, a0, false, b, (short)0, acc[0][t], false, false);
            acc[1][t] = __builtin_amdgcn_wmma_f32_16x16x32_f16(
                            false, a1, false, b, (short)0, acc[1][t], false, false);
        }
    }

    // Epilogue. D layout: vgpr r -> M = tile_base + 8*(lane>>4) + r, N = lane&15.
#pragma unroll
    for (int mi = 0; mi < 2; ++mi) {
        const int mbase = (tmg << 5) + (mi << 4) + ((lane >> 4) << 3);
#pragma unroll
        for (int t = 0; t < 4; ++t) {
            const int n = ncolt[t];
            if (n >= Ncols) continue;             // lane store guard only
#pragma unroll
            for (int r = 0; r < 8; ++r) {
                const int m = mbase + r;
                float v = acc[mi][t][r];
                if (bias)  v += bias[n];
                if (addin) v += addin[(size_t)m * (size_t)ldy + n];
                if (act == 1)      v = sigm(v);
                else if (act == 2) v = fmaxf(v, 0.0f);
                if (mul)   v *= mul[(size_t)m * (size_t)ldy + n];
                Y[(size_t)m * (size_t)ldy + n] = v;
            }
        }
    }
}

// ---------------------------------------------------------------------------
// Per-layer query scores: aq[b,i] = H[b,i,:] . wq + lb
// ---------------------------------------------------------------------------
__global__ void aq_kernel(const float* __restrict__ H, const float* __restrict__ wq,
                          const float* __restrict__ lb, float* __restrict__ aq)
{
    const int idx = blockIdx.x * blockDim.x + threadIdx.x;
    if (idx >= BB * NU) return;
    const float* r = H + (size_t)idx * HID;
    float s = lb[0];
    for (int k = 0; k < HID; ++k) s += r[k] * wq[k];
    aq[idx] = s;
}

// ---------------------------------------------------------------------------
// DAG attention step i: masked softmax over cached key scores, then
// M[b,:] = sum_j w_j * ( sm ? v0[b,j,:] : v1[b,j,:] ).  Step 0 -> M = 0.
// ---------------------------------------------------------------------------
__global__ void att_kernel(const float* __restrict__ aq, const float* __restrict__ ak,
                           const float* __restrict__ adj, const int* __restrict__ sm,
                           const float* __restrict__ v0, const float* __restrict__ v1,
                           float* __restrict__ Mb, int i)
{
    const int b = blockIdx.x, tid = threadIdx.x;
    __shared__ float w0[NU], w1[NU], red[128];
    if (i == 0) {
        for (int h = tid; h < HID; h += blockDim.x) Mb[b * HID + h] = 0.0f;
        return;
    }
    const float aqv = aq[b * NU + i];
    for (int j = tid; j < NU; j += blockDim.x) {
        const float mask = adj[((size_t)b * NU + i) * NU + j] * (j < i ? 1.0f : 0.0f);
        w0[j] = (aqv + ak[b * NU + j]) - (1.0f - mask) * 1e30f;
    }
    __syncthreads();
    float pm = -3.4e38f;
    for (int j = tid; j < NU; j += blockDim.x) pm = fmaxf(pm, w0[j]);
    red[tid] = pm; __syncthreads();
    for (int s = 64; s > 0; s >>= 1) { if (tid < s) red[tid] = fmaxf(red[tid], red[tid + s]); __syncthreads(); }
    const float mx = red[0];
    __syncthreads();
    float ps = 0.0f;
    for (int j = tid; j < NU; j += blockDim.x) { const float e = __expf(w0[j] - mx); w0[j] = e; ps += e; }
    red[tid] = ps; __syncthreads();
    for (int s = 64; s > 0; s >>= 1) { if (tid < s) red[tid] += red[tid + s]; __syncthreads(); }
    const float inv = 1.0f / red[0];
    __syncthreads();
    for (int j = tid; j < NU; j += blockDim.x) {
        const float w   = w0[j] * inv;
        const float sel = (float)sm[((size_t)b * NU + i) * NU + j];
        w0[j] = w * sel;
        w1[j] = w - w * sel;
    }
    __syncthreads();
    for (int h = tid; h < HID; h += blockDim.x) {
        float acc = 0.0f;
        for (int j = 0; j < NU; ++j) {
            const size_t o = ((size_t)b * NU + j) * HID + h;
            acc += w0[j] * v0[o] + w1[j] * v1[o];
        }
        Mb[b * HID + h] = acc;
    }
}

// ---------------------------------------------------------------------------
// GRU gating + H update + incremental key-score:
//   C = gru(x=h_i, h=M) via gic/ghc;  P = gru(x=M, h=h_i) via gip/ghp
//   Hout[b,i,:] = C + P;  ak[b,i] = (C+P) . wk
// ---------------------------------------------------------------------------
__global__ void gruelt_kernel(const float* __restrict__ gic, const float* __restrict__ ghc,
                              const float* __restrict__ gip, const float* __restrict__ ghp,
                              const float* __restrict__ Hin, float* __restrict__ Hout,
                              const float* __restrict__ Mb, const float* __restrict__ wk,
                              float* __restrict__ ak, int i)
{
    const int b = blockIdx.x, tid = threadIdx.x;
    __shared__ float red[128];
    float part = 0.0f;
    for (int h = tid; h < HID; h += blockDim.x) {
        const size_t rb = (size_t)b * 900;
        const float hi = Hin[(size_t)b * (NU * HID) + (size_t)i * HID + h];
        const float mv = Mb[b * HID + h];
        // C = _gru(h_i, M)
        float r = sigm(gic[rb + h] + ghc[rb + h]);
        float z = sigm(gic[rb + HID + h] + ghc[rb + HID + h]);
        float n = tanhf(gic[rb + 2 * HID + h] + r * ghc[rb + 2 * HID + h]);
        const float C = (1.0f - z) * n + z * mv;
        // P = _gru(M, h_i)
        r = sigm(gip[rb + h] + ghp[rb + h]);
        z = sigm(gip[rb + HID + h] + ghp[rb + HID + h]);
        n = tanhf(gip[rb + 2 * HID + h] + r * ghp[rb + 2 * HID + h]);
        const float P = (1.0f - z) * n + z * hi;
        const float nh = C + P;
        Hout[(size_t)b * (NU * HID) + (size_t)i * HID + h] = nh;
        part += nh * wk[h];
    }
    red[tid] = part; __syncthreads();
    for (int s = 64; s > 0; s >>= 1) { if (tid < s) red[tid] += red[tid + s]; __syncthreads(); }
    if (tid == 0) ak[b * NU + i] = red[0];
}

// ---------------------------------------------------------------------------
// Output head: logits = x2 @ w3^T + b3 ; softmax over 7 classes.
// ---------------------------------------------------------------------------
__global__ void out_kernel(const float* __restrict__ x2, const float* __restrict__ w3,
                           const float* __restrict__ b3, float* __restrict__ out)
{
    const int row = blockIdx.x * blockDim.x + threadIdx.x;
    if (row >= BB * NU) return;
    const float* xr = x2 + (size_t)row * HID;
    float acc[NCLS];
#pragma unroll
    for (int c = 0; c < NCLS; ++c) acc[c] = b3[c];
    for (int k = 0; k < HID; ++k) {
        const float xv = xr[k];
#pragma unroll
        for (int c = 0; c < NCLS; ++c) acc[c] += xv * w3[c * HID + k];
    }
    float mx = acc[0];
#pragma unroll
    for (int c = 1; c < NCLS; ++c) mx = fmaxf(mx, acc[c]);
    float s = 0.0f;
#pragma unroll
    for (int c = 0; c < NCLS; ++c) { acc[c] = __expf(acc[c] - mx); s += acc[c]; }
    const float inv = 1.0f / s;
#pragma unroll
    for (int c = 0; c < NCLS; ++c) out[(size_t)row * NCLS + c] = acc[c] * inv;
}

// ---------------------------------------------------------------------------
static inline void gemm(hipStream_t s, const float* X, int ldx, const float* W, int ldw,
                        const float* bias, const float* addin, const float* mul,
                        float* Y, int ldy, int M, int K, int N, int act)
{
    const int tilesN  = (N + 15) >> 4;
    const int groupsN = (tilesN + 3) >> 2;
    const int waves   = (M >> 5) * groupsN;       // 32 rows per wave
    const int blocks  = (waves + 7) / 8;
    wmma_gemm_kernel<<<blocks, 256, 0, s>>>(X, ldx, W, ldw, bias, addin, mul,
                                            Y, ldy, M, K, N, act);
}

extern "C" void kernel_launch(void* const* d_in, const int* in_sizes, int n_in,
                              void* d_out, int out_size, void* d_ws, size_t ws_size,
                              hipStream_t stream)
{
    (void)in_sizes; (void)n_in; (void)out_size; (void)ws_size;
    // --- inputs in setup_inputs() dict order (params flattened in insertion order)
    const float* ftext = (const float*)d_in[0];
    const float* faud  = (const float*)d_in[1];
    const float* adj   = (const float*)d_in[2];
    const int*   smask = (const int*)d_in[3];
    /* lengths d_in[4] unused (all == N) */
    const float* U1    = (const float*)d_in[5];
    const float* U2    = (const float*)d_in[6];
    const float* Pw    = (const float*)d_in[7];
    const float* Pb    = (const float*)d_in[8];
    const float* V1    = (const float*)d_in[9];
    const float* V2    = (const float*)d_in[10];
    const float* fc1w  = (const float*)d_in[11];
    const float* fc1b  = (const float*)d_in[12];
    const float* glw   = (const float*)d_in[13];   // (L,1,600)
    const float* glb   = (const float*)d_in[14];   // (L,)
    const float* Wr0   = (const float*)d_in[15];   // (L,300,300)
    const float* Wr1   = (const float*)d_in[16];
    const float* cwih  = (const float*)d_in[17];   // (L,900,300)
    const float* cwhh  = (const float*)d_in[18];
    const float* cbih  = (const float*)d_in[19];   // (L,900)
    const float* cbhh  = (const float*)d_in[20];
    const float* pwih  = (const float*)d_in[21];
    const float* pwhh  = (const float*)d_in[22];
    const float* pbih  = (const float*)d_in[23];
    const float* pbhh  = (const float*)d_in[24];
    const float* mw1   = (const float*)d_in[25];   // (300,1412)
    const float* mb1   = (const float*)d_in[26];
    const float* mw2   = (const float*)d_in[27];
    const float* mb2   = (const float*)d_in[28];
    const float* mw3   = (const float*)d_in[29];
    const float* mb3   = (const float*)d_in[30];
    float* out = (float*)d_out;
    float* ws  = (float*)d_ws;

    const size_t R  = (size_t)BB * NU;          // 12800 rows
    const size_t oFUSED = 0;
    const size_t oS1  = oFUSED + R * FOD;
    const size_t oH0  = oS1  + R * FOD;
    const size_t oH1A = oH0  + R * HID;
    const size_t oH1B = oH1A + R * HID;
    const size_t oV0  = oH1B + R * HID;
    const size_t oV1  = oV0  + R * HID;
    const size_t oTMP = oV1  + R * HID;
    const size_t oX2  = oTMP + R * HID;
    const size_t oAQ  = oX2  + R * HID;
    const size_t oAK  = oAQ  + R;
    const size_t oMB  = oAK  + R;
    const size_t oGIC = oMB  + (size_t)BB * HID;
    const size_t oGHC = oGIC + (size_t)BB * 900;
    const size_t oGIP = oGHC + (size_t)BB * 900;
    const size_t oGHP = oGIP + (size_t)BB * 900;

    float* FUSED = ws + oFUSED; float* S1 = ws + oS1;  float* H0 = ws + oH0;
    float* H1A = ws + oH1A;     float* H1B = ws + oH1B;
    float* V0B = ws + oV0;      float* V1B = ws + oV1;
    float* TMP = ws + oTMP;     float* X2 = ws + oX2;
    float* AQ = ws + oAQ;       float* AK = ws + oAK;  float* MB = ws + oMB;
    float* GIC = ws + oGIC; float* GHC = ws + oGHC; float* GIP = ws + oGIP; float* GHP = ws + oGHP;

    const int M = (int)R;

    // ---- bilinear pooling fusion (all WMMA) ----
    gemm(stream, ftext, TD, U1, TD, nullptr, nullptr, nullptr, S1,    FOD, M, TD,  FOD, 1); // S1=sig(T@U1^T)
    gemm(stream, faud,  TD, U2, TD, nullptr, nullptr, S1,      S1,    FOD, M, TD,  FOD, 1); // S1=sig(A@U2^T)*S1
    gemm(stream, ftext, TD, V1, TD, nullptr, nullptr, nullptr, FUSED, FOD, M, TD,  FOD, 0); // F=T@V1^T
    gemm(stream, faud,  TD, V2, TD, nullptr, FUSED,   nullptr, FUSED, FOD, M, TD,  FOD, 0); // F+=A@V2^T
    gemm(stream, S1,   FOD, Pw, FOD, Pb,     FUSED,   nullptr, FUSED, FOD, M, FOD, FOD, 0); // F=S1@Pw^T+Pb+F
    gemm(stream, FUSED, FOD, fc1w, FOD, fc1b, nullptr, nullptr, H0,   HID, M, FOD, HID, 2); // H0=relu(...)

    // ---- DAG-GNN layers (sequential scan with incremental V/key caches) ----
    for (int l = 0; l < LNUM; ++l) {
        const float* Hin  = (l == 0) ? H0  : H1A;
        float*       Hout = (l == 0) ? H1A : H1B;
        const float* wq = glw + (size_t)l * 2 * HID;
        const float* wk = wq + HID;
        const float* Wr0l = Wr0 + (size_t)l * HID * HID;
        const float* Wr1l = Wr1 + (size_t)l * HID * HID;
        const float* cwihl = cwih + (size_t)l * 3 * HID * HID;
        const float* cwhhl = cwhh + (size_t)l * 3 * HID * HID;
        const float* pwihl = pwih + (size_t)l * 3 * HID * HID;
        const float* pwhhl = pwhh + (size_t)l * 3 * HID * HID;
        const float* cbihl = cbih + (size_t)l * 3 * HID;
        const float* cbhhl = cbhh + (size_t)l * 3 * HID;
        const float* pbihl = pbih + (size_t)l * 3 * HID;
        const float* pbhhl = pbhh + (size_t)l * 3 * HID;

        hipMemsetAsync(V0B, 0, R * HID * sizeof(float), stream);
        hipMemsetAsync(V1B, 0, R * HID * sizeof(float), stream);
        hipMemsetAsync(AK,  0, R * sizeof(float), stream);
        aq_kernel<<<(int)((R + 255) / 256), 256, 0, stream>>>(Hin, wq, glb + l, AQ);

        for (int i = 0; i < NU; ++i) {
            att_kernel<<<BB, 128, 0, stream>>>(AQ, AK, adj, smask, V0B, V1B, MB, i);
            const float* hrow = Hin + (size_t)i * HID;                 // rows strided NU*HID
            gemm(stream, hrow, NU * HID, cwihl, HID, cbihl, nullptr, nullptr, GIC, 900, BB, HID, 900, 0);
            gemm(stream, MB,   HID,      cwhhl, HID, cbhhl, nullptr, nullptr, GHC, 900, BB, HID, 900, 0);
            gemm(stream, MB,   HID,      pwihl, HID, pbihl, nullptr, nullptr, GIP, 900, BB, HID, 900, 0);
            gemm(stream, hrow, NU * HID, pwhhl, HID, pbhhl, nullptr, nullptr, GHP, 900, BB, HID, 900, 0);
            gruelt_kernel<<<BB, 128, 0, stream>>>(GIC, GHC, GIP, GHP, Hin, Hout, MB, wk, AK, i);
            const float* nrow = Hout + (size_t)i * HID;
            gemm(stream, nrow, NU * HID, Wr0l, HID, nullptr, nullptr, nullptr, V0B + (size_t)i * HID, NU * HID, BB, HID, HID, 0);
            gemm(stream, nrow, NU * HID, Wr1l, HID, nullptr, nullptr, nullptr, V1B + (size_t)i * HID, NU * HID, BB, HID, HID, 0);
        }
    }

    // ---- MLP head: Hcat GEMM done as 4 column-slice GEMMs of mlp_w1 (ldw=1412) ----
    const int LD1 = HID * (LNUM + 1) + FOD;   // 1412
    gemm(stream, H0,    HID, mw1,         LD1, nullptr, nullptr, nullptr, TMP, HID, M, HID, HID, 0);
    gemm(stream, H1A,   HID, mw1 + HID,   LD1, nullptr, TMP,     nullptr, TMP, HID, M, HID, HID, 0);
    gemm(stream, H1B,   HID, mw1 + 2*HID, LD1, nullptr, TMP,     nullptr, TMP, HID, M, HID, HID, 0);
    gemm(stream, FUSED, FOD, mw1 + 3*HID, LD1, mb1,     TMP,     nullptr, TMP, HID, M, FOD, HID, 2); // x1=relu
    gemm(stream, TMP,   HID, mw2, HID, mb2, nullptr, nullptr, X2, HID, M, HID, HID, 2);              // x2=relu
    out_kernel<<<(int)((R + 127) / 128), 128, 0, stream>>>(X2, mw3, mb3, out);
}
